// XLAttention_15968688407240
// MI455X (gfx1250) — matrix operand order repaired and checked by reading, not compiled, not run on hardware
//
#include <hip/hip_runtime.h>
#include <hip/hip_bf16.h>

#define Bv   2
#define Tt   2048
#define Mm   2048
#define Cc   1024
#define Hh   16
#define Dd   64
#define TKk  4096   // Mm + Tt
#define BT   4096   // Bv * Tt
#define RELW 4096   // Tt + Mm

typedef __attribute__((ext_vector_type(16))) __bf16 v16bf;
typedef __attribute__((ext_vector_type(8)))  float  v8f;

union ABu { v16bf v; uint4 q[2]; };
union Bpk { v16bf v; unsigned u[8]; };

static __device__ __forceinline__ unsigned pack2bf(float a, float b) {
  unsigned short lo = __builtin_bit_cast(unsigned short, (__bf16)a);
  unsigned short hi = __builtin_bit_cast(unsigned short, (__bf16)b);
  return (unsigned)lo | ((unsigned)hi << 16);
}

static __device__ __forceinline__ v8f wmma_bf16(v16bf a, v16bf b, v8f c) {
  return __builtin_amdgcn_wmma_f32_16x16x32_bf16(false, a, false, b, (short)0, c, false, false);
}

// ---------------- conversion kernels ----------------

__global__ void k_convert_x(const float* __restrict__ x, __bf16* __restrict__ xb, int n) {
  int i = blockIdx.x * blockDim.x + threadIdx.x;
  if (i < n) xb[i] = (__bf16)x[i];
}

// W (K=1024 rows, N=1024 cols) fp32 -> Wt (N, K) bf16
__global__ void k_transpose_w(const float* __restrict__ W, __bf16* __restrict__ Wt) {
  int idx = blockIdx.x * blockDim.x + threadIdx.x;   // K*N threads
  int k = idx >> 10, n = idx & 1023;
  Wt[(size_t)n * Cc + k] = (__bf16)W[idx];
}

// xl_memory (B, M, 2, C): k -> k_b rows [0,M), v -> vT (B,H,D,TK) cols [0,M)
__global__ void k_convert_xl(const float* __restrict__ xl,
                             __bf16* __restrict__ kb, __bf16* __restrict__ vT) {
  int idx = blockIdx.x * blockDim.x + threadIdx.x;   // Bv*Mm*Cc threads
  int c = idx & (Cc - 1);
  int j = (idx >> 10) & (Mm - 1);
  int b = idx >> 21;
  size_t base = (((size_t)(b * Mm + j)) * 2) * Cc + c;
  kb[((size_t)(b * TKk + j)) * Cc + c] = (__bf16)xl[base];
  int h = c >> 6, d = c & 63;
  vT[(((size_t)(b * Hh + h)) * Dd + d) * TKk + j] = (__bf16)xl[base + Cc];
}

// ---------------- WMMA GEMM (16x64 tile / wave, K=1024, double-buffered) ----------------
// mode 0: q = x@Wq   -> bf16 row-major (BT, C)
// mode 1: k = x@Wk   -> bf16 k_b rows [M, M+T) + fp32 new_xl[...,0,:]
// mode 2: v = x@Wv   -> bf16 vT transposed + fp32 new_xl[...,1,:]
// mode 3: out = wv@Wp + bp -> fp32 d_out
__global__ __launch_bounds__(32) void k_gemm(
    const __bf16* __restrict__ A, const __bf16* __restrict__ Wt, int mode,
    __bf16* __restrict__ outB, float* __restrict__ outF, const float* __restrict__ bias) {
  const int lane = threadIdx.x;
  const int hl = lane >> 4, l15 = lane & 15;
  const int nbase = blockIdx.x * 64;        // 4 column tiles of 16
  const int rowbase = blockIdx.y * 16;
  const __bf16* arow  = A  + (size_t)(rowbase + l15) * Cc;
  const __bf16* brow0 = Wt + (size_t)(nbase  + l15) * Cc + 16 * hl;

  auto load_frag = [&](int kb0, ABu& a, ABu* b) {
    a.q[0] = *(const uint4*)(arow + kb0 + 16 * hl);       // A: lane=row, contiguous K16
    a.q[1] = *(const uint4*)(arow + kb0 + 16 * hl + 8);   // A: contiguous K16
#pragma unroll
    for (int nt = 0; nt < 4; nt++) {
      const __bf16* br = brow0 + (size_t)(16 * nt) * Cc + kb0;
      b[nt].q[0] = *(const uint4*)(br);                   // B: lane=col, contiguous K16
      b[nt].q[1] = *(const uint4*)(br + 8);
    }
  };

  v8f acc[4];
#pragma unroll
  for (int nt = 0; nt < 4; nt++) acc[nt] = (v8f){0.f,0.f,0.f,0.f,0.f,0.f,0.f,0.f};

  ABu a_cur, b_cur[4];
  load_frag(0, a_cur, b_cur);
  for (int kb0 = 0; kb0 < Cc; kb0 += 32) {
    // prefetch next k-step's fragments into a second register set (in flight
    // during this step's WMMAs -> partial s_wait_loadcnt instead of a drain)
    ABu a_nxt, b_nxt[4];
    const int kn = (kb0 + 32 < Cc) ? kb0 + 32 : kb0;
    load_frag(kn, a_nxt, b_nxt);
#pragma unroll
    for (int nt = 0; nt < 4; nt++)                        // reuse A across 4 B tiles
      acc[nt] = wmma_bf16(a_cur.v, b_cur[nt].v, acc[nt]);
    a_cur = a_nxt;
#pragma unroll
    for (int nt = 0; nt < 4; nt++) b_cur[nt] = b_nxt[nt];
  }

#pragma unroll
  for (int nt = 0; nt < 4; nt++) {
    const int c = nbase + nt * 16 + l15;
    if (mode == 0) {
#pragma unroll
      for (int r = 0; r < 8; r++) {
        int tok = rowbase + 8 * hl + r;
        outB[(size_t)tok * Cc + c] = (__bf16)acc[nt][r];
      }
    } else if (mode == 1) {
#pragma unroll
      for (int r = 0; r < 8; r++) {
        int tok = rowbase + 8 * hl + r;
        int b_ = tok >> 11, t = tok & (Tt - 1);
        outB[((size_t)(b_ * TKk + Mm + t)) * Cc + c] = (__bf16)acc[nt][r];
        outF[(((size_t)(b_ * Tt + t)) * 2 + 0) * Cc + c] = acc[nt][r];
      }
    } else if (mode == 2) {
      int h = c >> 6, d = c & 63;
#pragma unroll
      for (int r = 0; r < 8; r++) {
        int tok = rowbase + 8 * hl + r;
        int b_ = tok >> 11, t = tok & (Tt - 1);
        outB[(((size_t)(b_ * Hh + h)) * Dd + d) * TKk + Mm + t] = (__bf16)acc[nt][r];
        outF[(((size_t)(b_ * Tt + t)) * 2 + 1) * Cc + c] = acc[nt][r];
      }
    } else {
      float bb = bias[c];
#pragma unroll
      for (int r = 0; r < 8; r++) {
        int tok = rowbase + 8 * hl + r;
        outF[(size_t)tok * Cc + c] = acc[nt][r] + bb;
      }
    }
  }
}

// ---------------- flash attention (transposed scores), 1 wave / 16 queries ----------------
__global__ __launch_bounds__(32) void k_attn(
    const __bf16* __restrict__ qb, const __bf16* __restrict__ kbm,
    const __bf16* __restrict__ vT, const float* __restrict__ rel,
    __bf16* __restrict__ wv) {
  const int lane = threadIdx.x;
  const int hl = lane >> 4, l15 = lane & 15;
  const int qbase = blockIdx.x * 16;
  const int bh = blockIdx.y;
  const int b = bh / Hh, h = bh % Hh;
  const int iq = qbase + l15;           // this lane's query column
  const int jlim = iq + Mm;             // causal: keep j <= iq + M

  // Q^T as B operand (lane = query column, table-order K per half), reused for all keys
  const __bf16* qrow = qb + (size_t)(b * Tt + iq) * Cc + h * Dd;
  ABu Bq0, Bq1;
  Bq0.q[0] = *(const uint4*)(qrow + 8 * hl);
  Bq0.q[1] = *(const uint4*)(qrow + 16 + 8 * hl);
  Bq1.q[0] = *(const uint4*)(qrow + 32 + 8 * hl);
  Bq1.q[1] = *(const uint4*)(qrow + 48 + 8 * hl);

  v8f o[4];
#pragma unroll
  for (int t = 0; t < 4; t++) o[t] = (v8f){0.f,0.f,0.f,0.f,0.f,0.f,0.f,0.f};
  float mrun = -3.0e38f, srun = 0.f;

  const float* relrow = rel + ((size_t)h * Tt + iq) * RELW;
  const int jend = qbase + 15 + Mm + 1;   // exclusive j bound for this tile

  for (int jb0 = 0; jb0 < jend; jb0 += 32) {
    const __bf16* kr0 = kbm + (size_t)(b * TKk + jb0 + l15) * Cc + h * Dd;
    const __bf16* kr1 = kr0 + (size_t)16 * Cc;
    const float*  rp  = relrow + jb0 + 8 * hl;
    __builtin_prefetch(rp + 32, 0, 1);          // global_prefetch_b8 for next step

    // ---- issue ALL loads for this step up front: K frags, V frags, rel ----
    ABu a00, a01, a10, a11;
    a00.q[0] = *(const uint4*)(kr0 + 8 * hl);
    a00.q[1] = *(const uint4*)(kr0 + 16 + 8 * hl);
    a01.q[0] = *(const uint4*)(kr0 + 32 + 8 * hl);
    a01.q[1] = *(const uint4*)(kr0 + 48 + 8 * hl);
    a10.q[0] = *(const uint4*)(kr1 + 8 * hl);
    a10.q[1] = *(const uint4*)(kr1 + 16 + 8 * hl);
    a11.q[0] = *(const uint4*)(kr1 + 32 + 8 * hl);
    a11.q[1] = *(const uint4*)(kr1 + 48 + 8 * hl);

    const __bf16* vbase = vT + (((size_t)(b * Hh + h)) * Dd + l15) * TKk + jb0;
    ABu av[4];
#pragma unroll
    for (int dt = 0; dt < 4; dt++) {            // V frags retire behind softmax VALU
      av[dt].q[0] = *(const uint4*)(vbase + 16 * hl);
      av[dt].q[1] = *(const uint4*)(vbase + 16 * hl + 8);
      vbase += (size_t)16 * TKk;
    }

    float relv0[8], relv1[8];
    *(float4*)&relv0[0] = *(const float4*)(rp);
    *(float4*)&relv0[4] = *(const float4*)(rp + 4);
    *(float4*)&relv1[0] = *(const float4*)(rp + 16);
    *(float4*)&relv1[4] = *(const float4*)(rp + 20);

    // ---- scores S^T = K_tile @ Q^T (waits only on K frags) ----
    v8f s0 = {0.f,0.f,0.f,0.f,0.f,0.f,0.f,0.f};
    v8f s1 = {0.f,0.f,0.f,0.f,0.f,0.f,0.f,0.f};
    s0 = wmma_bf16(a00.v, Bq0.v, s0);
    s0 = wmma_bf16(a01.v, Bq1.v, s0);
    s1 = wmma_bf16(a10.v, Bq0.v, s1);
    s1 = wmma_bf16(a11.v, Bq1.v, s1);

    // ---- rel-pos add, scale 1/sqrt(64), causal mask ----
    const int j0 = jb0 + 8 * hl;       // tile0 j for r=0
    const int j1 = j0 + 16;            // tile1 j for r=0
    float w0[8], w1[8];
#pragma unroll
    for (int r = 0; r < 8; r++) {
      float a0 = (s0[r] + relv0[r]) * 0.125f;
      float a1 = (s1[r] + relv1[r]) * 0.125f;
      w0[r] = (j0 + r > jlim) ? -3.0e38f : a0;
      w1[r] = (j1 + r > jlim) ? -3.0e38f : a1;
    }

    // ---- online softmax over this lane's query column ----
    float mx = w0[0];
#pragma unroll
    for (int r = 0; r < 8; r++) { mx = fmaxf(mx, w0[r]); mx = fmaxf(mx, w1[r]); }
    mx = fmaxf(mx, __shfl_xor(mx, 16));
    float mnew = fmaxf(mrun, mx);
    float alpha = __expf(mrun - mnew);
    srun *= alpha;
#pragma unroll
    for (int t = 0; t < 4; t++) o[t] *= alpha;
    mrun = mnew;

    float p0[8], p1[8], ss = 0.f;
#pragma unroll
    for (int r = 0; r < 8; r++) {
      p0[r] = __expf(w0[r] - mnew);
      p1[r] = __expf(w1[r] - mnew);
      ss += p0[r] + p1[r];
    }
    ss += __shfl_xor(ss, 16);
    srun += ss;

    // ---- P^T -> B operand (half-wave shuffle only, no LDS) ----
    unsigned t0p[4], t1p[4];
#pragma unroll
    for (int i = 0; i < 4; i++) {
      t0p[i] = pack2bf(p0[2 * i], p0[2 * i + 1]);
      t1p[i] = pack2bf(p1[2 * i], p1[2 * i + 1]);
    }
    Bpk Bp;
#pragma unroll
    for (int i = 0; i < 4; i++) {
      unsigned o0p = __shfl_xor(t0p[i], 16);
      unsigned o1p = __shfl_xor(t1p[i], 16);
      if (hl == 0) { Bp.u[i] = t0p[i]; Bp.u[4 + i] = o0p; }
      else         { Bp.u[i] = o1p;    Bp.u[4 + i] = t1p[i]; }
    }

    // ---- O^T += V^T_tile @ P^T (V frags already resident) ----
#pragma unroll
    for (int dt = 0; dt < 4; dt++)
      o[dt] = wmma_bf16(av[dt].v, Bp.v, o[dt]);
  }

  // ---- normalize and write wv (bf16, row-major (BT, C)) ----
  float inv = 1.0f / srun;
  __bf16* orow = wv + (size_t)(b * Tt + iq) * Cc + h * Dd + 8 * hl;
#pragma unroll
  for (int dt = 0; dt < 4; dt++) {
#pragma unroll
    for (int r = 0; r < 8; r++)
      orow[dt * 16 + r] = (__bf16)(o[dt][r] * inv);
  }
}

// ---------------- launch ----------------
extern "C" void kernel_launch(void* const* d_in, const int* in_sizes, int n_in,
                              void* d_out, int out_size, void* d_ws, size_t ws_size,
                              hipStream_t stream) {
  const float* rel = (const float*)d_in[0];
  const float* x   = (const float*)d_in[1];
  const float* xl  = (const float*)d_in[2];
  const float* Wq  = (const float*)d_in[3];
  const float* Wk  = (const float*)d_in[4];
  const float* Wv  = (const float*)d_in[5];
  const float* Wp  = (const float*)d_in[6];
  const float* bp  = (const float*)d_in[7];

  float* out_f  = (float*)d_out;                       // (B,T,C)
  float* xl_out = out_f + (size_t)BT * Cc;             // (B,T,2,C)

  char* ws = (char*)d_ws;
  size_t off = 0;
  auto alloc = [&](size_t bytes) { void* p = ws + off; off += (bytes + 255) & ~(size_t)255; return p; };
  __bf16* x_b  = (__bf16*)alloc((size_t)BT * Cc * 2);
  __bf16* q_b  = (__bf16*)alloc((size_t)BT * Cc * 2);
  __bf16* k_b  = (__bf16*)alloc((size_t)Bv * TKk * Cc * 2);
  __bf16* v_t  = (__bf16*)alloc((size_t)Bv * TKk * Cc * 2);
  __bf16* wv_b = (__bf16*)alloc((size_t)BT * Cc * 2);
  __bf16* Wqt  = (__bf16*)alloc((size_t)Cc * Cc * 2);
  __bf16* Wkt  = (__bf16*)alloc((size_t)Cc * Cc * 2);
  __bf16* Wvt  = (__bf16*)alloc((size_t)Cc * Cc * 2);
  __bf16* Wpt  = (__bf16*)alloc((size_t)Cc * Cc * 2);

  k_convert_x<<<(BT * Cc) / 256, 256, 0, stream>>>(x, x_b, BT * Cc);
  k_transpose_w<<<(Cc * Cc) / 256, 256, 0, stream>>>(Wq, Wqt);
  k_transpose_w<<<(Cc * Cc) / 256, 256, 0, stream>>>(Wk, Wkt);
  k_transpose_w<<<(Cc * Cc) / 256, 256, 0, stream>>>(Wv, Wvt);
  k_transpose_w<<<(Cc * Cc) / 256, 256, 0, stream>>>(Wp, Wpt);
  k_convert_xl<<<(Bv * Mm * Cc) / 256, 256, 0, stream>>>(xl, k_b, v_t);

  dim3 gg(Cc / 64, BT / 16);
  k_gemm<<<gg, 32, 0, stream>>>(x_b, Wqt, 0, q_b, nullptr, nullptr);
  k_gemm<<<gg, 32, 0, stream>>>(x_b, Wkt, 1, k_b, xl_out, nullptr);
  k_gemm<<<gg, 32, 0, stream>>>(x_b, Wvt, 2, v_t, xl_out, nullptr);

  k_attn<<<dim3(Tt / 16, Bv * Hh), 32, 0, stream>>>(q_b, k_b, v_t, rel, wv_b);

  k_gemm<<<gg, 32, 0, stream>>>(wv_b, Wpt, 3, nullptr, out_f, bp);
}
